// SimplePCNet_34643206209621
// MI455X (gfx1250) — matrix operand order repaired and checked by reading, not compiled, not run on hardware
//
#include <hip/hip_runtime.h>
#include <hip/hip_bf16.h>
#include <hip/hip_fp16.h>

#define GRID 256
#define HBITS 21
#define HSLOTS (1u << HBITS)
#define HMASK  (HSLOTS - 1u)
#define NPB 1024   // pooling partial blocks

typedef __attribute__((ext_vector_type(16))) _Float16 v16h;
typedef __attribute__((ext_vector_type(8)))  _Float16 v8h;
typedef __attribute__((ext_vector_type(8)))  float    v8f;

__device__ __forceinline__ unsigned hash_key(int key) {
    return (((unsigned)key) * 2654435761u) >> (32 - HBITS);
}

__global__ void fill_i32(int* __restrict__ p, int n, int v) {
    int i = blockIdx.x * blockDim.x + threadIdx.x;
    if (i < n) p[i] = v;
}

// Insert (key -> row index) into open-addressing table. Keys are unique.
__global__ void build_hash(const int* __restrict__ coords, const int* __restrict__ bidx,
                           int n, int* __restrict__ hk, int* __restrict__ hv) {
    int i = blockIdx.x * blockDim.x + threadIdx.x;
    if (i >= n) return;
    int key = (((bidx[i] << 8) | coords[i * 3 + 0]) << 8 | coords[i * 3 + 1]) << 8 | coords[i * 3 + 2];
    unsigned h = hash_key(key);
    for (;;) {
        int old = atomicCAS(&hk[h], -1, key);
        if (old == -1 || old == key) { hv[h] = i; return; }
        h = (h + 1) & HMASK;
    }
}

__device__ __forceinline__ int hash_lookup(const int* __restrict__ hk,
                                           const int* __restrict__ hv, int key) {
    unsigned h = hash_key(key);
    for (;;) {
        int k = hk[h];
        if (k == key) return hv[h];
        if (k == -1)  return -1;
        h = (h + 1) & HMASK;
    }
}

// Pre-pack W[27][Cin][32] (fp32) into a WMMA B lane layout (contiguous k = 16*half + j, as gather_a), f16.
// Layout: wpk[k][nhalf][lane][j], j=0..15 ; element = W[k][ K=(lane>>4)*16+j ][ N=nhalf*16+(lane&15) ]
__global__ void pack_w(const float* __restrict__ W, _Float16* __restrict__ out, int Cin) {
    int gid = blockIdx.x * blockDim.x + threadIdx.x;
    if (gid >= 27 * 2 * 32) return;
    int k    = gid >> 6;
    int nh   = (gid >> 5) & 1;
    int lane = gid & 31;
    int n     = nh * 16 + (lane & 15);
    int kbase = (lane >> 4) * 16;
    for (int j = 0; j < 16; ++j) {
        int kk = kbase + j;
        float v = (kk < Cin) ? W[(size_t)(k * Cin + kk) * 32 + n] : 0.f;
        out[(size_t)((k * 2 + nh) * 32 + lane) * 16 + j] = (_Float16)v;
    }
}

// Gather one A-operand row-slice in contiguous k order (k = 16*half + j, as pack_w).
// lanes 0-15: K{0-15}; lanes 16-31: K{16-31}
template <bool CIN4>
__device__ __forceinline__ v16h gather_a(const _Float16* __restrict__ inh,
                                         const float* __restrict__ inf4,
                                         int idx, int hlf) {
    v16h a;
    #pragma unroll
    for (int j = 0; j < 16; ++j) a[j] = (_Float16)0.f;
    if (idx >= 0) {
        if constexpr (CIN4) {        // Cin = 4, fp32 input: only K<4 nonzero (lanes 0-15, elems 0-3)
            if (hlf == 0) {
                #pragma unroll
                for (int j = 0; j < 4; ++j) a[j] = (_Float16)inf4[(size_t)idx * 4 + j];
            }
        } else {                     // Cin = 32, f16 input: two 16B loads
            v8h lo = *(const v8h*)(inh + (size_t)idx * 32 + hlf * 16);
            v8h hi = *(const v8h*)(inh + (size_t)idx * 32 + hlf * 16 + 8);
            #pragma unroll
            for (int j = 0; j < 8; ++j) { a[j] = lo[j]; a[8 + j] = hi[j]; }
        }
    }
    return a;
}

// Gather + WMMA sparse conv. One wave per 32-row output tile (two 16-row A tiles),
// 32 output channels. Each lane does ONE hash lookup (row tb+lane); per-tile row
// indices are broadcast with ds_bpermute. Per offset: 2 B loads feed 4 WMMAs.
template <bool CIN4>
__global__ __launch_bounds__(128) void sparse_conv_wmma(
    const _Float16* __restrict__ inh, const float* __restrict__ inf4,
    const int* __restrict__ ocoords, const int* __restrict__ obidx, int M,
    const int* __restrict__ hk, const int* __restrict__ hv,
    const _Float16* __restrict__ wpk, int stride,
    _Float16* __restrict__ outh)
{
    const int lane = threadIdx.x & 31;
    const int wave = threadIdx.x >> 5;
    const int tb   = (blockIdx.x * 4 + wave) * 32;   // 32 rows per wave
    const int r    = lane & 15;
    const int hlf  = lane >> 4;
    const int lrow = tb + lane;                      // this lane's lookup row
    const bool lok = lrow < M;

    int cx = 0, cy = 0, cz = 0, cb = 0;
    if (lok) {
        cx = ocoords[lrow * 3 + 0];
        cy = ocoords[lrow * 3 + 1];
        cz = ocoords[lrow * 3 + 2];
        cb = obidx[lrow];
    }

    v8f c00 = {0.f, 0.f, 0.f, 0.f, 0.f, 0.f, 0.f, 0.f};
    v8f c01 = c00, c10 = c00, c11 = c00;

    for (int k = 0; k < 27; ++k) {
        int dx = (k / 9 - 1) * stride;
        int dy = ((k / 3) % 3 - 1) * stride;
        int dz = (k % 3 - 1) * stride;

        // one hash probe per output row (all 32 lanes unique)
        int idx = -1;
        if (lok) {
            int qx = cx + dx, qy = cy + dy, qz = cz + dz;
            if (((qx | qy | qz) >= 0) && qx < GRID && qy < GRID && qz < GRID) {
                int key = (((cb << 8) | qx) << 8 | qy) << 8 | qz;
                idx = hash_lookup(hk, hv, key);
            }
        }
        // broadcast per-tile row indices: tile0 rows = lanes 0-15, tile1 rows = lanes 16-31
        int idx0 = __builtin_amdgcn_ds_bpermute(r << 2, idx);
        int idx1 = __builtin_amdgcn_ds_bpermute((16 + r) << 2, idx);

        v16h a0 = gather_a<CIN4>(inh, inf4, idx0, hlf);
        v16h a1 = gather_a<CIN4>(inh, inf4, idx1, hlf);

        // B operands: pre-packed, 32 bytes per lane per N-half, shared by both A tiles
        const v8h* b0p = (const v8h*)(wpk + (size_t)((k * 2 + 0) * 32 + lane) * 16);
        const v8h* b1p = (const v8h*)(wpk + (size_t)((k * 2 + 1) * 32 + lane) * 16);
        v8h b0lo = b0p[0], b0hi = b0p[1];
        v8h b1lo = b1p[0], b1hi = b1p[1];
        v16h b0, b1;
        #pragma unroll
        for (int j = 0; j < 8; ++j) {
            b0[j] = b0lo[j]; b0[8 + j] = b0hi[j];
            b1[j] = b1lo[j]; b1[8 + j] = b1hi[j];
        }

        c00 = __builtin_amdgcn_wmma_f32_16x16x32_f16(false, a0, false, b0, (short)0, c00, false, false);
        c01 = __builtin_amdgcn_wmma_f32_16x16x32_f16(false, a0, false, b1, (short)0, c01, false, false);
        c10 = __builtin_amdgcn_wmma_f32_16x16x32_f16(false, a1, false, b0, (short)0, c10, false, false);
        c11 = __builtin_amdgcn_wmma_f32_16x16x32_f16(false, a1, false, b1, (short)0, c11, false, false);
    }

    // C/D layout: VGPR j -> row j + 8*hlf (tile-local), col = r (cX0) / 16+r (cX1)
    #pragma unroll
    for (int j = 0; j < 8; ++j) {
        int sr0 = tb + hlf * 8 + j;
        if (sr0 < M) {
            outh[(size_t)sr0 * 32 + r]      = (_Float16)c00[j];
            outh[(size_t)sr0 * 32 + 16 + r] = (_Float16)c01[j];
        }
        int sr1 = tb + 16 + hlf * 8 + j;
        if (sr1 < M) {
            outh[(size_t)sr1 * 32 + r]      = (_Float16)c10[j];
            outh[(size_t)sr1 * 32 + 16 + r] = (_Float16)c11[j];
        }
    }
}

// Deterministic pooling stage 1: one wave per contiguous row chunk; b2 is sorted,
// so within a chunk each batch is one run -> flush-on-change, no atomics.
__global__ void pool_partial(const _Float16* __restrict__ z, const int* __restrict__ bseg,
                             int M2, float* __restrict__ partials, int* __restrict__ pcounts) {
    int blk = blockIdx.x;
    int t   = threadIdx.x;   // 32 threads, one channel each
    for (int b = 0; b < 8; ++b) partials[(size_t)(blk * 8 + b) * 32 + t] = 0.f;
    if (t == 0) for (int b = 0; b < 8; ++b) pcounts[blk * 8 + b] = 0;

    int chunk = (M2 + NPB - 1) / NPB;
    int r0 = blk * chunk;
    int r1 = r0 + chunk; if (r1 > M2) r1 = M2;

    float acc = 0.f; int curb = -1; int cnt = 0;
    for (int r = r0; r < r1; ++r) {
        int bb = bseg[r];
        if (bb != curb) {
            if (curb >= 0) {
                partials[(size_t)(blk * 8 + curb) * 32 + t] = acc;
                if (t == 0) pcounts[blk * 8 + curb] = cnt;
            }
            curb = bb; acc = 0.f; cnt = 0;
        }
        acc += (float)z[(size_t)r * 32 + t];
        ++cnt;
    }
    if (curb >= 0) {
        partials[(size_t)(blk * 8 + curb) * 32 + t] = acc;
        if (t == 0) pcounts[blk * 8 + curb] = cnt;
    }
}

// Deterministic pooling stage 2: sequential reduction over partials.
__global__ void pool_final(const float* __restrict__ partials, const int* __restrict__ pcounts,
                           float* __restrict__ out) {
    int t = threadIdx.x;
    if (t >= 256) return;
    int b = t >> 5, ch = t & 31;
    float s = 0.f, c = 0.f;
    for (int blk = 0; blk < NPB; ++blk) {
        s += partials[(size_t)(blk * 8 + b) * 32 + ch];
        c += (float)pcounts[blk * 8 + b];
    }
    out[b * 32 + ch] = s / c;
}

extern "C" void kernel_launch(void* const* d_in, const int* in_sizes, int n_in,
                              void* d_out, int out_size, void* d_ws, size_t ws_size,
                              hipStream_t stream) {
    const float* feats  = (const float*)d_in[0];
    const float* W1     = (const float*)d_in[1];
    const float* W2     = (const float*)d_in[2];
    const float* W3     = (const float*)d_in[3];
    const int* coords   = (const int*)d_in[4];
    const int* bidx     = (const int*)d_in[5];
    const int* coords2  = (const int*)d_in[6];
    const int* bseg2    = (const int*)d_in[7];
    const int N1 = in_sizes[5];
    const int M2 = in_sizes[7];

    char* ws = (char*)d_ws;
    size_t off = 0;
    auto wsalloc = [&](size_t bytes) -> void* {
        void* p = ws + off;
        off = (off + bytes + 255) & ~(size_t)255;
        return p;
    };
    int* h1k = (int*)wsalloc((size_t)HSLOTS * sizeof(int));
    int* h1v = (int*)wsalloc((size_t)HSLOTS * sizeof(int));
    int* h2k = (int*)wsalloc((size_t)HSLOTS * sizeof(int));
    int* h2v = (int*)wsalloc((size_t)HSLOTS * sizeof(int));
    const size_t wpk_elems = (size_t)27 * 2 * 32 * 16;
    _Float16* wp1 = (_Float16*)wsalloc(wpk_elems * sizeof(_Float16));
    _Float16* wp2 = (_Float16*)wsalloc(wpk_elems * sizeof(_Float16));
    _Float16* wp3 = (_Float16*)wsalloc(wpk_elems * sizeof(_Float16));
    _Float16* x1h = (_Float16*)wsalloc((size_t)N1 * 32 * sizeof(_Float16));
    _Float16* y2h = (_Float16*)wsalloc((size_t)M2 * 32 * sizeof(_Float16));
    float* partials = (float*)wsalloc((size_t)NPB * 8 * 32 * sizeof(float));
    int*   pcounts  = (int*)wsalloc((size_t)NPB * 8 * sizeof(int));
    _Float16* z3h = x1h;   // x1 dead after conv2: alias

    // Hash tables
    fill_i32<<<(HSLOTS + 255) / 256, 256, 0, stream>>>(h1k, HSLOTS, -1);
    fill_i32<<<(HSLOTS + 255) / 256, 256, 0, stream>>>(h2k, HSLOTS, -1);
    build_hash<<<(N1 + 255) / 256, 256, 0, stream>>>(coords,  bidx,  N1, h1k, h1v);
    build_hash<<<(M2 + 255) / 256, 256, 0, stream>>>(coords2, bseg2, M2, h2k, h2v);

    // Weight packs (B-matrix layout, f16)
    pack_w<<<7, 256, 0, stream>>>(W1, wp1, 4);
    pack_w<<<7, 256, 0, stream>>>(W2, wp2, 32);
    pack_w<<<7, 256, 0, stream>>>(W3, wp3, 32);

    // conv1: submanifold, Cin=4 fp32 -> x1h
    int tiles1 = (N1 + 31) / 32;
    sparse_conv_wmma<true><<<(tiles1 + 3) / 4, 128, 0, stream>>>(
        nullptr, feats, coords, bidx, N1, h1k, h1v, wp1, 1, x1h);

    // conv2: stride-2 output set, gathers from stride-1 grid -> y2h
    int tiles2 = (M2 + 31) / 32;
    sparse_conv_wmma<false><<<(tiles2 + 3) / 4, 128, 0, stream>>>(
        x1h, nullptr, coords2, bseg2, M2, h1k, h1v, wp2, 1, y2h);

    // conv3: submanifold at tensor stride 2 -> z3h
    sparse_conv_wmma<false><<<(tiles2 + 3) / 4, 128, 0, stream>>>(
        y2h, nullptr, coords2, bseg2, M2, h2k, h2v, wp3, 2, z3h);

    // Global average pool per batch -> d_out [8,32]
    pool_partial<<<NPB, 32, 0, stream>>>(z3h, bseg2, M2, partials, pcounts);
    pool_final<<<1, 256, 0, stream>>>(partials, pcounts, (float*)d_out);
}
